// GNN_15762529976411
// MI455X (gfx1250) — matrix operand order repaired and checked by reading, not compiled, not run on hardware
//
#include <hip/hip_runtime.h>

// GCN: B=4, N=2048, F=128.  3 layers of  h = act(Anorm @ h @ W + b) + h.
// MI455X (gfx1250, wave32, WMMA) strategy:
//   1) deg row-sum -> dinv = (deg+eps)^-1/2
//   2) Abf = bf16(dinv_i * adj * dinv_j)      (ONCE; 32MB, L2-resident all layers)
//   3) Wt  = bf16(W^T) x3
//   per layer:
//   4) Ht  = bf16(h^T) [B][F][N]              (tiled transpose -> contiguous B-frags)
//   5) fused layer kernel: per WG, 32-row x 128-col output tile, 8 waves.
//      Wave w owns column tile w and BOTH row tiles -> B-fragment shared by
//      2 WMMAs (3 B128 loads per v_wmma instead of 4).
//      GEMM1: 2 x 64 WMMAs over K=N; Z -> LDS (bf16, 8KB) re-layout;
//      GEMM2: 2 x 4 WMMAs over K=F (W-frag shared across row tiles);
//      epilogue: +bias, ReLU (layers 1,2), +residual, f32 store.

#define BB 4
#define NN 2048
#define FF 128

typedef __attribute__((ext_vector_type(16))) __bf16          bf16x16;
typedef __attribute__((ext_vector_type(8)))  float           f32x8;
typedef __attribute__((ext_vector_type(8)))  unsigned short  u16x8;

union BF16Frag {
  bf16x16 v;
  u16x8   h[2];
};

__device__ __forceinline__ unsigned short f2bf(float f) {
  unsigned int u = __float_as_uint(f);
  u += 0x7FFFu + ((u >> 16) & 1u);   // round-to-nearest-even
  return (unsigned short)(u >> 16);
}

// ---------------------------------------------------------------- degree ----
__global__ __launch_bounds__(256) void deg_kernel(const float* __restrict__ adj,
                                                  float* __restrict__ dinv) {
  __shared__ float red[256];
  const int i = blockIdx.x;
  const int b = blockIdx.y;
  const float* row = adj + ((size_t)b * NN + i) * NN;
  float s = 0.f;
  for (int j = threadIdx.x; j < NN; j += 256) s += row[j];
  red[threadIdx.x] = s;
  __syncthreads();
  for (int off = 128; off > 0; off >>= 1) {
    if (threadIdx.x < off) red[threadIdx.x] += red[threadIdx.x + off];
    __syncthreads();
  }
  if (threadIdx.x == 0) dinv[(size_t)b * NN + i] = 1.0f / sqrtf(red[0] + 1e-6f);
}

// -------------------------------------------- normalize adj -> bf16 ---------
__global__ __launch_bounds__(256) void norm_kernel(const float* __restrict__ adj,
                                                   const float* __restrict__ dinv,
                                                   unsigned short* __restrict__ Abf) {
  size_t idx = (size_t)blockIdx.x * 256 + threadIdx.x;
  const size_t total = (size_t)BB * NN * NN;
  if (idx >= total) return;
  size_t bi = idx >> 11;        // / NN
  int    j  = (int)(idx & (NN - 1));
  int    b  = (int)(bi >> 11);
  int    i  = (int)(bi & (NN - 1));
  float v = dinv[b * NN + i] * adj[idx] * dinv[b * NN + j];
  Abf[idx] = f2bf(v);
}

// ------------------------------------------ W -> bf16(W^T) ------------------
__global__ __launch_bounds__(256) void wt_kernel(const float* __restrict__ W,
                                                 unsigned short* __restrict__ Wt) {
  int idx = blockIdx.x * 256 + threadIdx.x;   // 16384 elements
  int f = idx >> 7;
  int o = idx & (FF - 1);
  Wt[o * FF + f] = f2bf(W[f * FF + o]);
}

// ------------------------------------- h (f32 [B][N][F]) -> bf16 [B][F][N] --
__global__ __launch_bounds__(256) void ht_kernel(const float* __restrict__ h,
                                                 unsigned short* __restrict__ Ht) {
  __shared__ float s[32][33];
  const int b  = blockIdx.z;
  const int n0 = blockIdx.y * 32;
  const int f0 = blockIdx.x * 32;
  const int tx = threadIdx.x;   // 0..31
  const int ty = threadIdx.y;   // 0..7
  const float* hb = h + (size_t)b * NN * FF;
#pragma unroll
  for (int i = 0; i < 4; ++i) {
    int n = n0 + ty + i * 8;
    s[ty + i * 8][tx] = hb[(size_t)n * FF + f0 + tx];
  }
  __syncthreads();
  unsigned short* ob = Ht + (size_t)b * FF * NN;
#pragma unroll
  for (int i = 0; i < 4; ++i) {
    int f = f0 + ty + i * 8;
    ob[(size_t)f * NN + n0 + tx] = f2bf(s[tx][ty + i * 8]);
  }
}

// --------------------------------------------------- fused GCN layer --------
__global__ __launch_bounds__(256) void gcn_layer(
    const unsigned short* __restrict__ Abf,   // [B][N][N] bf16 (normalized adj)
    const unsigned short* __restrict__ Ht,    // [B][F][N] bf16 (h transposed)
    const unsigned short* __restrict__ Wt,    // [F][F] bf16   (W transposed, o-major)
    const float* __restrict__ bias,           // [F]
    const float* __restrict__ h_in,           // [B][N][F] f32 (residual source)
    float* __restrict__ h_out,                // [B][N][F] f32
    int relu) {
  __shared__ unsigned short zlds[32 * FF];    // 8KB Z tile (bf16)

  const int tid  = threadIdx.x;
  const int lane = tid & 31;
  const int wid  = tid >> 5;        // wave id 0..7 -> output column tile
  const int ln16 = lane & 15;
  const int hi   = lane >> 4;       // lane half select
  const int row0 = blockIdx.x * 32; // node-row super-tile (2 x 16 rows)
  const int b    = blockIdx.y;

  const unsigned short* A = Abf + (size_t)b * NN * NN;
  const unsigned short* H = Ht + (size_t)b * FF * NN;

  // ---- GEMM1: Z[32 x 16(wid)] = A[row0:+32, :] x H[:, wid*16:+16] -------
  // A-frag (16x32): lane ln16 = row; lanes 0-15 K={kk..+7, kk+16..+23},
  //                 lanes 16-31 K={kk+8..+15, kk+24..+31}
  // B-frag (32x16): lane ln16 = col; lanes 0-15 K={kk..+7, kk+16..+23},
  //                 lanes 16-31 K={kk+8..+15, kk+24..+31} -- SHARED by 2 WMMAs
  f32x8 acc0 = {}, acc1 = {};
  const unsigned short* arow0 = A + (size_t)(row0 + ln16) * NN + hi * 8;
  const unsigned short* arow1 = arow0 + (size_t)16 * NN;
  const unsigned short* hcol  = H + (size_t)(wid * 16 + ln16) * NN + hi * 8;
#pragma unroll 2
  for (int kk = 0; kk < NN; kk += 32) {
    BF16Frag a0, a1, bf;
    a0.h[0] = *(const u16x8*)(arow0 + kk);
    a0.h[1] = *(const u16x8*)(arow0 + kk + 16);
    a1.h[0] = *(const u16x8*)(arow1 + kk);
    a1.h[1] = *(const u16x8*)(arow1 + kk + 16);
    bf.h[0] = *(const u16x8*)(hcol + kk);
    bf.h[1] = *(const u16x8*)(hcol + kk + 16);
    acc0 = __builtin_amdgcn_wmma_f32_16x16x32_bf16(false, a0.v, false, bf.v,
                                                   (short)0, acc0, false, false);
    acc1 = __builtin_amdgcn_wmma_f32_16x16x32_bf16(false, a1.v, false, bf.v,
                                                   (short)0, acc1, false, false);
  }

  // ---- spill both Z row-tiles to LDS as bf16 (row-major 32 x 128) -------
  {
    const int c = wid * 16 + ln16;
#pragma unroll
    for (int r = 0; r < 8; ++r) {
      int m = r + hi * 8;   // D-frag layout: VGPR r holds rows r / r+8
      zlds[m * FF + c]        = f2bf(acc0[r]);
      zlds[(m + 16) * FF + c] = f2bf(acc1[r]);
    }
  }
  __syncthreads();

  // ---- GEMM2: Out[32 x 16(wid)] = Z[32 x 128] x Wt cols -----------------
  f32x8 o0 = {}, o1 = {};
  const unsigned short* wcol  = Wt + (size_t)(wid * 16 + ln16) * FF + hi * 8;
  const unsigned short* zrow0 = &zlds[ln16 * FF + hi * 8];
  const unsigned short* zrow1 = &zlds[(16 + ln16) * FF + hi * 8];
#pragma unroll
  for (int kt = 0; kt < FF; kt += 32) {
    BF16Frag a0, a1, bf;
    a0.h[0] = *(const u16x8*)(zrow0 + kt);
    a0.h[1] = *(const u16x8*)(zrow0 + kt + 16);
    a1.h[0] = *(const u16x8*)(zrow1 + kt);
    a1.h[1] = *(const u16x8*)(zrow1 + kt + 16);
    bf.h[0] = *(const u16x8*)(wcol + kt);
    bf.h[1] = *(const u16x8*)(wcol + kt + 16);
    o0 = __builtin_amdgcn_wmma_f32_16x16x32_bf16(false, a0.v, false, bf.v,
                                                 (short)0, o0, false, false);
    o1 = __builtin_amdgcn_wmma_f32_16x16x32_bf16(false, a1.v, false, bf.v,
                                                 (short)0, o1, false, false);
  }

  // ---- epilogue: +bias, optional ReLU, +residual, f32 store -------------
  const int   c  = wid * 16 + ln16;
  const float bv = bias[c];
  const float* hin  = h_in  + ((size_t)b * NN + row0) * FF;
  float*       hout = h_out + ((size_t)b * NN + row0) * FF;
#pragma unroll
  for (int r = 0; r < 8; ++r) {
    int m = r + hi * 8;
    float v0 = o0[r] + bv;
    float v1 = o1[r] + bv;
    if (relu) { v0 = fmaxf(v0, 0.0f); v1 = fmaxf(v1, 0.0f); }
    v0 += hin[m * FF + c];
    v1 += hin[(m + 16) * FF + c];
    hout[m * FF + c]        = v0;
    hout[(m + 16) * FF + c] = v1;
  }
}

// ---------------------------------------------------------------------------
extern "C" void kernel_launch(void* const* d_in, const int* in_sizes, int n_in,
                              void* d_out, int out_size, void* d_ws, size_t ws_size,
                              hipStream_t stream) {
  const float* x   = (const float*)d_in[0];
  const float* adj = (const float*)d_in[1];
  const float* W1  = (const float*)d_in[2];
  const float* b1  = (const float*)d_in[3];
  const float* W2  = (const float*)d_in[4];
  const float* b2  = (const float*)d_in[5];
  const float* W3  = (const float*)d_in[6];
  const float* b3  = (const float*)d_in[7];
  float* out = (float*)d_out;

  // workspace carve-up (~44.2 MB total, all chunks 16B aligned)
  char* ws = (char*)d_ws;
  unsigned short* Abf  = (unsigned short*)ws; ws += (size_t)BB * NN * NN * 2; // 32 MB
  float*          dinv = (float*)ws;          ws += (size_t)BB * NN * 4;      // 32 KB
  unsigned short* Wt   = (unsigned short*)ws; ws += (size_t)3 * FF * FF * 2;  // 96 KB
  unsigned short* Ht   = (unsigned short*)ws; ws += (size_t)BB * FF * NN * 2; // 2 MB
  float*          hA   = (float*)ws;          ws += (size_t)BB * NN * FF * 4; // 4 MB
  float*          hB   = (float*)ws;                                          // 4 MB

  // 1) degree -> dinv
  deg_kernel<<<dim3(NN, BB), 256, 0, stream>>>(adj, dinv);
  // 2) normalized adjacency in bf16 (once; reused for all 3 layers)
  {
    size_t total = (size_t)BB * NN * NN;
    norm_kernel<<<(unsigned)((total + 255) / 256), 256, 0, stream>>>(adj, dinv, Abf);
  }
  // 3) transposed bf16 weights
  wt_kernel<<<64, 256, 0, stream>>>(W1, Wt);
  wt_kernel<<<64, 256, 0, stream>>>(W2, Wt + FF * FF);
  wt_kernel<<<64, 256, 0, stream>>>(W3, Wt + 2 * FF * FF);

  const dim3 htg(FF / 32, NN / 32, BB), htb(32, 8);
  const dim3 lg(NN / 32, BB);

  // layer 1: ReLU + residual
  ht_kernel<<<htg, htb, 0, stream>>>(x, Ht);
  gcn_layer<<<lg, 256, 0, stream>>>(Abf, Ht, Wt, b1, x, hA, 1);
  // layer 2: ReLU + residual
  ht_kernel<<<htg, htb, 0, stream>>>(hA, Ht);
  gcn_layer<<<lg, 256, 0, stream>>>(Abf, Ht, Wt + FF * FF, b2, hA, hB, 1);
  // layer 3: no activation + residual
  ht_kernel<<<htg, htb, 0, stream>>>(hB, Ht);
  gcn_layer<<<lg, 256, 0, stream>>>(Abf, Ht, Wt + 2 * FF * FF, b3, hB, out, 0);
}
